// DirGCNConv_43611097924219
// MI455X (gfx1250) — compile-verified, matrix operand order checked by reading, not run on hardware
//
#include <hip/hip_runtime.h>
#include <hip/hip_bf16.h>

typedef __attribute__((ext_vector_type(16))) _Float16 v16h;
typedef __attribute__((ext_vector_type(8)))  float    v8f;

#define NNODES 50000
#define NEDGES 625000
#define DIM    128
#define ALPHA  0.5f

// ---------------------------------------------------------------------------
// 1) zero the f32 workspace region (degrees + both aggregation buffers)
// ---------------------------------------------------------------------------
__global__ void zero_kernel(float* p, int n) {
    int i = blockIdx.x * blockDim.x + threadIdx.x;
    if (i < n) p[i] = 0.0f;
}

// ---------------------------------------------------------------------------
// 2) degree computation: atomic scatter of ones
// ---------------------------------------------------------------------------
__global__ void deg_kernel(const int* __restrict__ row, const int* __restrict__ col,
                           float* __restrict__ out_deg, float* __restrict__ in_deg, int e) {
    int i = blockIdx.x * blockDim.x + threadIdx.x;
    if (i < e) {
        atomicAdd(&out_deg[row[i]], 1.0f);
        atomicAdd(&in_deg[col[i]], 1.0f);
    }
}

// ---------------------------------------------------------------------------
// 3) deg -> deg^{-1/2} (masked), in place
// ---------------------------------------------------------------------------
__global__ void inv_kernel(float* __restrict__ out_deg, float* __restrict__ in_deg, int n) {
    int i = blockIdx.x * blockDim.x + threadIdx.x;
    if (i < n) {
        float od = out_deg[i];
        float id = in_deg[i];
        out_deg[i] = od > 0.0f ? rsqrtf(od) : 0.0f;
        in_deg[i]  = id > 0.0f ? rsqrtf(id) : 0.0f;
    }
}

// ---------------------------------------------------------------------------
// 4) convert W_sd / W_ds to f16 (L2-resident staging) + combined bias
// ---------------------------------------------------------------------------
__global__ void wcvt_kernel(const float* __restrict__ Wsd, const float* __restrict__ Wds,
                            const float* __restrict__ bsd, const float* __restrict__ bds,
                            _Float16* __restrict__ Wsd_h, _Float16* __restrict__ Wds_h,
                            float* __restrict__ bias_c) {
    int i = blockIdx.x * blockDim.x + threadIdx.x;
    if (i < DIM * DIM) {
        Wsd_h[i] = (_Float16)Wsd[i];
        Wds_h[i] = (_Float16)Wds[i];
    }
    if (i < DIM) {
        bias_c[i] = ALPHA * bsd[i] + (1.0f - ALPHA) * bds[i];
    }
}

// ---------------------------------------------------------------------------
// 5) edge scatter: one wave32 per edge, one float4 (4 feats) per lane.
//    alpha / (1-alpha) folded into the scatter weights so the GEMM epilogue
//    can run a single accumulator chain. Atomics resolve in L2 (agg ~51MB).
// ---------------------------------------------------------------------------
__global__ void scatter_kernel(const float* __restrict__ x,
                               const int* __restrict__ row, const int* __restrict__ col,
                               const float* __restrict__ out_inv, const float* __restrict__ in_inv,
                               float* __restrict__ agg_sd, float* __restrict__ agg_ds, int e) {
    int gtid = blockIdx.x * blockDim.x + threadIdx.x;
    int edge = gtid >> 5;
    int lane = gtid & 31;
    if (edge >= e) return;

    int r = row[edge];
    int c = col[edge];
    float w  = out_inv[r] * in_inv[c];
    float wa = w * ALPHA;           // scale for sd branch
    float wb = w * (1.0f - ALPHA);  // scale for ds branch

    const float4 xd = ((const float4*)(x + (size_t)c * DIM))[lane]; // x[col] lane quad
    const float4 xs = ((const float4*)(x + (size_t)r * DIM))[lane]; // x[row] lane quad

    float* asd = agg_sd + (size_t)r * DIM + lane * 4; // msg_fwd -> row
    float* ads = agg_ds + (size_t)c * DIM + lane * 4; // msg_bwd -> col
    atomicAdd(asd + 0, wa * xd.x);
    atomicAdd(asd + 1, wa * xd.y);
    atomicAdd(asd + 2, wa * xd.z);
    atomicAdd(asd + 3, wa * xd.w);
    atomicAdd(ads + 0, wb * xs.x);
    atomicAdd(ads + 1, wb * xs.y);
    atomicAdd(ads + 2, wb * xs.z);
    atomicAdd(ads + 3, wb * xs.w);
}

// ---------------------------------------------------------------------------
// 6) WMMA GEMM epilogue: out = agg_sd_scaled @ Wsd^T + agg_ds_scaled @ Wds^T + bias_c
//    One wave per 16x16 output tile; 8 x v_wmma_f32_16x16x32_f16 per tile.
//
//    ISA 16-bit A layout (16x32, MxK): lane L holds row M=L&15;
//      lanes 0-15:  K = {0..7} in halves 0..7,  K = {16..23} in halves 8..15
//      lanes 16-31: K = {8..15},                K = {24..31}
//    B (32x16) mirrored: lane holds column n = L&15 == row j of W (row-major,
//    contiguous in k since out = agg @ W^T).
// ---------------------------------------------------------------------------
__device__ __forceinline__ v16h load_a_frag_f32(const float* __restrict__ base,
                                                int k0, int lane) {
    // base = agg + m0*DIM ; returns fragment for rows m0..m0+15, K slice [k0,k0+32)
    int r  = lane & 15;
    int kh = (lane >> 4) << 3;  // 0 or 8
    const float* p = base + (size_t)r * DIM + k0 + kh;
    v16h f;
#pragma unroll
    for (int i = 0; i < 8; ++i) f[i] = (_Float16)p[i];
#pragma unroll
    for (int i = 0; i < 8; ++i) f[i + 8] = (_Float16)p[i + 16];
    return f;
}

__device__ __forceinline__ v16h load_b_frag_h(const _Float16* __restrict__ W,
                                              int j0, int k0, int lane) {
    int r  = j0 + (lane & 15);
    int kh = (lane >> 4) << 3;
    const _Float16* p = W + (size_t)r * DIM + k0 + kh;
    v16h f;
#pragma unroll
    for (int i = 0; i < 8; ++i) f[i] = p[i];
#pragma unroll
    for (int i = 0; i < 8; ++i) f[i + 8] = p[i + 16];
    return f;
}

__global__ void gemm_kernel(const float* __restrict__ agg_sd, const float* __restrict__ agg_ds,
                            const _Float16* __restrict__ Wsd_h, const _Float16* __restrict__ Wds_h,
                            const float* __restrict__ bias_c, float* __restrict__ out) {
    int m0   = blockIdx.x * 16;           // 3125 blocks * 16 = 50000 rows exactly
    int wave = threadIdx.x >> 5;          // 8 waves -> 8 column tiles (128/16)
    int lane = threadIdx.x & 31;
    int j0   = wave * 16;

    v8f acc = {};
#pragma unroll
    for (int k0 = 0; k0 < DIM; k0 += 32) {
        v16h a = load_a_frag_f32(agg_sd + (size_t)m0 * DIM, k0, lane);
        v16h b = load_b_frag_h(Wsd_h, j0, k0, lane);
        acc = __builtin_amdgcn_wmma_f32_16x16x32_f16(false, a, false, b,
                                                     (short)0, acc, false, false);
    }
#pragma unroll
    for (int k0 = 0; k0 < DIM; k0 += 32) {
        v16h a = load_a_frag_f32(agg_ds + (size_t)m0 * DIM, k0, lane);
        v16h b = load_b_frag_h(Wds_h, j0, k0, lane);
        acc = __builtin_amdgcn_wmma_f32_16x16x32_f16(false, a, false, b,
                                                     (short)0, acc, false, false);
    }

    // C/D layout: VGPR v -> (M = v + 8*(lane>=16), N = lane&15)
    int n  = j0 + (lane & 15);
    int mo = (lane >> 4) << 3;
    float bc = bias_c[n];
#pragma unroll
    for (int v = 0; v < 8; ++v)
        out[(size_t)(m0 + mo + v) * DIM + n] = acc[v] + bc;
}

// ---------------------------------------------------------------------------
// launch
// ---------------------------------------------------------------------------
extern "C" void kernel_launch(void* const* d_in, const int* in_sizes, int n_in,
                              void* d_out, int out_size, void* d_ws, size_t ws_size,
                              hipStream_t stream) {
    const float* x    = (const float*)d_in[0];
    const int*   ei   = (const int*)d_in[1];   // (2, E) flat: row then col
    const float* Wsd  = (const float*)d_in[2];
    const float* bsd  = (const float*)d_in[3];
    const float* Wds  = (const float*)d_in[4];
    const float* bds  = (const float*)d_in[5];
    float*       out  = (float*)d_out;

    const int* row = ei;
    const int* col = ei + NEDGES;

    // workspace layout (f32 region first, 16B-aligned f16 region after)
    float* ws       = (float*)d_ws;
    float* out_inv  = ws;                               // N   (deg then inv, in place)
    float* in_inv   = ws + NNODES;                      // N
    float* agg_sd   = ws + 2 * NNODES;                  // N*D
    float* agg_ds   = agg_sd + (size_t)NNODES * DIM;    // N*D
    _Float16* Wsd_h = (_Float16*)(agg_ds + (size_t)NNODES * DIM);
    _Float16* Wds_h = Wsd_h + DIM * DIM;
    float* bias_c   = (float*)(Wds_h + DIM * DIM);

    const int zero_n = 2 * NNODES + 2 * NNODES * DIM;

    zero_kernel<<<(zero_n + 255) / 256, 256, 0, stream>>>(ws, zero_n);

    deg_kernel<<<(NEDGES + 255) / 256, 256, 0, stream>>>(row, col, out_inv, in_inv, NEDGES);

    inv_kernel<<<(NNODES + 255) / 256, 256, 0, stream>>>(out_inv, in_inv, NNODES);

    wcvt_kernel<<<(DIM * DIM + 255) / 256, 256, 0, stream>>>(Wsd, Wds, bsd, bds,
                                                             Wsd_h, Wds_h, bias_c);

    // one wave32 per edge
    const long long sc_threads = (long long)NEDGES * 32;
    scatter_kernel<<<(unsigned)((sc_threads + 255) / 256), 256, 0, stream>>>(
        x, row, col, out_inv, in_inv, agg_sd, agg_ds, NEDGES);

    // 50000/16 = 3125 row tiles; 8 waves/block cover the 8 column tiles
    gemm_kernel<<<NNODES / 16, 256, 0, stream>>>(agg_sd, agg_ds, Wsd_h, Wds_h, bias_c, out);
}
